// DeepSEA_concatenation_36447092474111
// MI455X (gfx1250) — matrix operand order repaired and checked by reading, not compiled, not run on hardware
//
#include <hip/hip_runtime.h>

// ---------------------------------------------------------------------------
// Types for CDNA5 WMMA / TDM (gfx1250, wave32)
// ---------------------------------------------------------------------------
typedef __bf16 bf16;
typedef __attribute__((ext_vector_type(16))) __bf16 v16bf;
typedef __attribute__((ext_vector_type(8)))  __bf16 v8bf;
typedef __attribute__((ext_vector_type(8)))  float  v8f;
typedef __attribute__((ext_vector_type(4))) unsigned int u32x4;
typedef __attribute__((ext_vector_type(8))) int          i32x8;
typedef __attribute__((ext_vector_type(4))) int          i32x4;

#ifndef __has_builtin
#define __has_builtin(x) 0
#endif
#if __has_builtin(__builtin_amdgcn_tensor_load_to_lds) && \
    __has_builtin(__builtin_amdgcn_s_wait_tensorcnt)
#define USE_TDM 1
#else
#define USE_TDM 0
#endif

#define N_NODES 20000
#define N_EDGES 320000

#define TM 128
#define TN 128
#define TK 32
#define LDSS 40   // padded LDS stride (halfwords): conflict-free fragment reads

#if USE_TDM
// ---------------------------------------------------------------------------
// Tensor Data Mover: DMA one A tile (rows x 32 bf16) global -> LDS.
// D# per ISA 8.3/8.4: 2D tile 32 x 128, data_size=2B, pad every 16 DWORDs
// (=1 row of 32 bf16) by 4 DWORDs (=8 bf16)  ->  LDS row stride = 40 bf16.
// OOB reads (past tensor_dim0/1) return zero == our zero-padding semantics.
// Issued by one wave; EXEC is ignored by TENSOR ops.
// ---------------------------------------------------------------------------
__device__ __forceinline__ void tdm_load_tile_A(const bf16* gsrc, unsigned lds_off,
                                                unsigned rem_k, unsigned rem_m,
                                                unsigned lda)
{
    unsigned long long ga = (unsigned long long)(const void*)gsrc;
    u32x4 g0;
    g0[0] = 1u;                                       // count=1, user descriptor
    g0[1] = lds_off;                                  // lds_addr (bytes)
    g0[2] = (unsigned)(ga & 0xFFFFFFFFu);             // global_addr lo
    g0[3] = (unsigned)((ga >> 32) & 0x01FFFFFFu)      // global_addr hi (57b)
          | (2u << 30);                               // type=2 ("image")
    i32x8 g1;
    unsigned long long st0 = (unsigned long long)lda; // tensor_dim0_stride
    g1[0] = (int)((1u << 16)        // data_size = 1 -> 2 bytes
                | (1u << 20)        // pad_enable
                | (3u << 22)        // pad_interval = 3 -> every 16 DWORDs
                | (3u << 25));      // pad_amount   = 3 -> 4 DWORDs
    g1[1] = (int)((rem_k & 0xFFFFu) << 16);                   // tensor_dim0 lo16
    g1[2] = (int)((rem_k >> 16) | ((rem_m & 0xFFFFu) << 16)); // td0 hi / td1 lo
    g1[3] = (int)((rem_m >> 16) | (32u << 16));               // td1 hi / tile_dim0=32
    g1[4] = (int)128;                                         // tile_dim1 = 128
    g1[5] = (int)(st0 & 0xFFFFFFFFu);                         // stride lo
    g1[6] = (int)((st0 >> 32) & 0xFFFFu);                     // stride hi
    g1[7] = 0;
    i32x4 z4 = {0, 0, 0, 0};
#if __clang_major__ >= 23
    i32x8 z8 = {0, 0, 0, 0, 0, 0, 0, 0};
    __builtin_amdgcn_tensor_load_to_lds(g0, g1, z4, z4, z8, 0);
#else
    __builtin_amdgcn_tensor_load_to_lds(g0, g1, z4, z4, 0);
#endif
}
#endif // USE_TDM

// ---------------------------------------------------------------------------
// Generic bf16 x bf16 -> f32 GEMM using v_wmma_f32_16x16x32_bf16
// C[M,N] = act( A[M,K] @ B[K,N] + bias )   (all row-major)
// bias_mode: 0 none, 1 per-row (M), 2 per-col (N); act: 0 none, 1 relu, 2 sigmoid
// Block tile 128x128, 8 waves; each wave: 64x32 = 4x2 fragments.
// ---------------------------------------------------------------------------
__global__ __launch_bounds__(256)
void gemm_bf16_wmma_k(const bf16* __restrict__ A, const bf16* __restrict__ Bm,
                      float* __restrict__ C,
                      int M, int N, int K, int lda, int ldb, int ldc,
                      const float* __restrict__ bias, int bias_mode, int act)
{
    __shared__ __align__(16) bf16 sA[TM * LDSS];   // [m][k]
    __shared__ __align__(16) bf16 sB[TN * LDSS];   // [n][k]  (B transposed in LDS)

    const int tid  = threadIdx.x;
    const int lane = tid & 31;
    const int wid  = tid >> 5;
    const int wm   = wid >> 2;      // 0..1  (64 rows each)
    const int wn   = wid & 3;       // 0..3  (32 cols each)
    const int half = lane >> 4;     // 0/1
    const int l16  = lane & 15;

    const int mb = blockIdx.y * TM;
    const int nb = blockIdx.x * TN;

#if USE_TDM
    const unsigned ldsA_off = (unsigned)(unsigned long long)(void*)&sA[0];
#endif

    v8f acc[8] = {};

    const int kTiles = (K + TK - 1) / TK;
    for (int kt = 0; kt < kTiles; ++kt) {
        const int k0 = kt * TK;

        // --- stage A tile (TM x TK) ---
#if USE_TDM
        if (wid == 0)
            tdm_load_tile_A(A + (long long)mb * lda + k0, ldsA_off,
                            (unsigned)(K - k0), (unsigned)(M - mb), (unsigned)lda);
#else
        for (int i = tid; i < TM * TK; i += 256) {
            int m  = i >> 5;
            int k  = i & 31;
            int gm = mb + m, gk = k0 + k;
            int cm = gm < M ? gm : (M - 1);       // clamp: unconditional load
            int ck = gk < K ? gk : (K - 1);
            bf16 v = A[(long long)cm * lda + ck];
            if (gm >= M || gk >= K) v = (bf16)0.0f;
            sA[m * LDSS + k] = v;
        }
#endif
        // --- stage B tile (TK x TN) transposed into [n][k], clamped loads ---
        for (int i = tid; i < TK * TN; i += 256) {
            int k  = i >> 7;
            int n  = i & 127;
            int gk = k0 + k, gn = nb + n;
            int ck = gk < K ? gk : (K - 1);
            int cn = gn < N ? gn : (N - 1);
            bf16 v = Bm[(long long)ck * ldb + cn];
            if (gk >= K || gn >= N) v = (bf16)0.0f;
            sB[n * LDSS + k] = v;
        }
        // prefetch next B tile rows (global_prefetch_b8)
        if (kt + 1 < kTiles) {
            int gk2 = k0 + TK + (tid >> 3);
            int gn2 = nb + (tid & 7) * 16;
            if (gk2 < K && gn2 < N)
                __builtin_prefetch(&Bm[(long long)gk2 * ldb + gn2], 0, 1);
        }
#if USE_TDM
        if (wid == 0) __builtin_amdgcn_s_wait_tensorcnt(0);
#endif
        __syncthreads();

        // --- fragment loads (ISA 7.12.2 layouts) ---
        v16bf af[4], bfr[2];
#pragma unroll
        for (int fm = 0; fm < 4; ++fm) {
            // A 16x32: lane m=l16, K = {half*8+0..7, 16+half*8+0..7}
            const bf16* p = &sA[(wm * 64 + fm * 16 + l16) * LDSS + half * 8];
            v8bf lo = *(const v8bf*)(p);
            v8bf hi = *(const v8bf*)(p + 16);
            af[fm] = __builtin_shufflevector(lo, hi,
                        0,1,2,3,4,5,6,7,8,9,10,11,12,13,14,15);
        }
#pragma unroll
        for (int fn = 0; fn < 2; ++fn) {
            // B 32x16: lane n=l16, K = {half*8+0..7, 16+half*8+0..7}
            const bf16* p = &sB[(wn * 32 + fn * 16 + l16) * LDSS + half * 8];
            v8bf lo = *(const v8bf*)(p);
            v8bf hi = *(const v8bf*)(p + 16);
            bfr[fn] = __builtin_shufflevector(lo, hi,
                        0,1,2,3,4,5,6,7,8,9,10,11,12,13,14,15);
        }
#pragma unroll
        for (int fm = 0; fm < 4; ++fm)
#pragma unroll
            for (int fn = 0; fn < 2; ++fn)
                acc[fm * 2 + fn] = __builtin_amdgcn_wmma_f32_16x16x32_bf16(
                    false, af[fm], false, bfr[fn],
                    (short)0, acc[fm * 2 + fn], false, false);
        __syncthreads();
    }

    // --- epilogue: C/D layout: lane l16 = n, rows m = half*8 + v ---
#pragma unroll
    for (int fm = 0; fm < 4; ++fm) {
#pragma unroll
        for (int fn = 0; fn < 2; ++fn) {
            int n = nb + wn * 32 + fn * 16 + l16;
            if (n >= N) continue;
            v8f a = acc[fm * 2 + fn];
#pragma unroll
            for (int v = 0; v < 8; ++v) {
                int m = mb + wm * 64 + fm * 16 + half * 8 + v;
                if (m >= M) continue;
                float x = a[v];
                if (bias_mode == 1)      x += bias[m];
                else if (bias_mode == 2) x += bias[n];
                if (act == 1)      x = fmaxf(x, 0.0f);
                else if (act == 2) x = 1.0f / (1.0f + __expf(-x));
                C[(long long)m * ldc + n] = x;
            }
        }
    }
}

// ---------------------------------------------------------------------------
// Elementwise / layout kernels
// ---------------------------------------------------------------------------
__global__ void f32_to_bf16_k(const float* __restrict__ in, bf16* __restrict__ out,
                              long long n)
{
    long long i = (long long)blockIdx.x * blockDim.x + threadIdx.x;
    if (i < n) out[i] = (bf16)in[i];
}

// in: R x C row-major (f32)  ->  out: C x R row-major (bf16)   (for W^T)
__global__ void f32_to_bf16_T_k(const float* __restrict__ in, bf16* __restrict__ out,
                                int R, int C)
{
    long long i = (long long)blockIdx.x * blockDim.x + threadIdx.x;
    long long total = (long long)R * C;
    if (i >= total) return;
    int r = (int)(i / C);
    int c = (int)(i % C);
    out[(long long)c * R + r] = (bf16)in[i];
}

__global__ void zero_f32_k(float* __restrict__ p, long long n)
{
    long long i = (long long)blockIdx.x * blockDim.x + threadIdx.x;
    if (i < n) p[i] = 0.0f;
}

// im2col for 1D VALID conv: out[(ci*KW+k)][b*Lout+l] = x[b][ci][l+k]  (bf16)
template <typename T>
__global__ void im2col_k(const T* __restrict__ x, bf16* __restrict__ out,
                         int B, int Cin, int Lin, int KW, int Lout)
{
    long long total = (long long)Cin * KW * B * Lout;
    long long i = (long long)blockIdx.x * blockDim.x + threadIdx.x;
    if (i >= total) return;
    int ncols = B * Lout;
    int col  = (int)(i % ncols);
    int rowk = (int)(i / ncols);
    int k  = rowk % KW;
    int ci = rowk / KW;
    int l  = col % Lout;
    int b  = col / Lout;
    float v = (float)x[((long long)b * Cin + ci) * Lin + l + k];
    out[i] = (bf16)v;
}

// relu + maxpool4 + transpose (co, b*L) -> (b, co, Lp), bf16 out
__global__ void relu_pool4_k(const float* __restrict__ C, bf16* __restrict__ P,
                             int B, int Co, int Lout, int Lp)
{
    long long total = (long long)B * Co * Lp;
    long long i = (long long)blockIdx.x * blockDim.x + threadIdx.x;
    if (i >= total) return;
    int lp = (int)(i % Lp);
    int t  = (int)(i / Lp);
    int co = t % Co;
    int b  = t / Co;
    long long base = (long long)co * ((long long)B * Lout) + (long long)b * Lout + 4 * lp;
    float m = C[base];
    m = fmaxf(m, C[base + 1]);
    m = fmaxf(m, C[base + 2]);
    m = fmaxf(m, C[base + 3]);
    m = fmaxf(m, 0.0f);
    P[((long long)b * Co + co) * Lp + lp] = (bf16)m;
}

// conv3 output (co, b*53) -> Y[b][co*53+l] with relu, bf16
__global__ void pack_y_k(const float* __restrict__ C3, bf16* __restrict__ Y)
{
    long long total = 64LL * 960 * 53;
    long long i = (long long)blockIdx.x * blockDim.x + threadIdx.x;
    if (i >= total) return;
    int l  = (int)(i % 53);
    int t  = (int)(i / 53);
    int co = t % 960;
    int b  = t / 960;
    float v = fmaxf(C3[(long long)co * (64 * 53) + (long long)b * 53 + l], 0.0f);
    Y[(long long)b * 50880 + co * 53 + l] = (bf16)v;
}

// one wave per node: s_row = h.a[0:F], s_col = h.a[F:2F]
__global__ void gat_scores_k(const float* __restrict__ h, const float* __restrict__ a,
                             int F, float* __restrict__ srow, float* __restrict__ scol,
                             int nn)
{
    int wid  = threadIdx.x >> 5;
    int lane = threadIdx.x & 31;
    int node = blockIdx.x * 8 + wid;
    if (node >= nn) return;
    float r = 0.0f, c = 0.0f;
    for (int f = lane; f < F; f += 32) {
        float hv = h[(long long)node * F + f];
        r += hv * a[f];
        c += hv * a[F + f];
    }
#pragma unroll
    for (int off = 16; off > 0; off >>= 1) {
        r += __shfl_xor(r, off, 32);
        c += __shfl_xor(c, off, 32);
    }
    if (lane == 0) { srow[node] = r; scol[node] = c; }
}

// per-edge attention weight e = exp(-leakyrelu(s_row[r]+s_col[c], 0.2)); denom += e
__global__ void gat_edge_k(const int* __restrict__ er, const int* __restrict__ ec,
                           const float* __restrict__ srow, const float* __restrict__ scol,
                           float* __restrict__ e, float* __restrict__ denom, int E)
{
    int i = blockIdx.x * blockDim.x + threadIdx.x;
    if (i >= E) return;
    int r = er[i], c = ec[i];
    float x  = srow[r] + scol[c];
    float lr = (x >= 0.0f) ? x : 0.2f * x;
    float ev = __expf(-lr);
    e[i] = ev;
    atomicAdd(&denom[r], ev);
}

// num[r][f] += e[edge] * h[c][f]   (edge x feature scatter-add)
__global__ void gat_scatter_k(const int* __restrict__ er, const int* __restrict__ ec,
                              const float* __restrict__ e, const float* __restrict__ h,
                              float* __restrict__ num, int E, int F)
{
    long long total = (long long)E * F;
    long long i = (long long)blockIdx.x * blockDim.x + threadIdx.x;
    if (i >= total) return;
    int f  = (int)(i % F);
    int ei = (int)(i / F);
    int r = er[ei], c = ec[ei];
    atomicAdd(&num[(long long)r * F + f], e[ei] * h[(long long)c * F + f]);
}

// out = elu(num / (denom + eps)); write bf16 into concat slot [node][off + f]
__global__ void gat_finalize_k(const float* __restrict__ num, const float* __restrict__ denom,
                               bf16* __restrict__ xout, int nn, int F, int ld, int off)
{
    long long total = (long long)nn * F;
    long long i = (long long)blockIdx.x * blockDim.x + threadIdx.x;
    if (i >= total) return;
    int f    = (int)(i % F);
    int node = (int)(i / F);
    float o = num[i] / (denom[node] + 1e-16f);
    o = (o > 0.0f) ? o : (__expf(o) - 1.0f);
    xout[(long long)node * ld + off + f] = (bf16)o;
}

// cat[b][j] = j<925 ? seq_out[b][j] : relu(struct[b][j-925])   (bf16)
__global__ void pack_cat_k(const float* __restrict__ seqo, const float* __restrict__ str,
                           bf16* __restrict__ cat)
{
    long long total = 64LL * 1053;
    long long i = (long long)blockIdx.x * blockDim.x + threadIdx.x;
    if (i >= total) return;
    int j = (int)(i % 1053);
    int b = (int)(i / 1053);
    float v = (j < 925) ? seqo[(long long)b * 925 + j]
                        : fmaxf(str[(long long)b * 128 + (j - 925)], 0.0f);
    cat[i] = (bf16)v;
}

// ---------------------------------------------------------------------------
// Host driver
// ---------------------------------------------------------------------------
static inline dim3 g1d(long long n) { return dim3((unsigned)((n + 255) / 256)); }

extern "C" void kernel_launch(void* const* d_in, const int* in_sizes, int n_in,
                              void* d_out, int out_size, void* d_ws, size_t ws_size,
                              hipStream_t stream)
{
    (void)in_sizes; (void)n_in; (void)out_size; (void)ws_size;

    // ---- inputs (setup_inputs order) ----
    const float* seq_in   = (const float*)d_in[0];   // (64,4,1000)
    const float* node_in  = (const float*)d_in[1];   // (20000,768)
    const float* index_in = (const float*)d_in[2];   // (64,20000)
    const int*   e_row    = (const int*)  d_in[3];   // (320000)
    const int*   e_col    = (const int*)  d_in[4];   // (320000)
    const float* c1w = (const float*)d_in[5];  const float* c1b = (const float*)d_in[6];
    const float* c2w = (const float*)d_in[7];  const float* c2b = (const float*)d_in[8];
    const float* c3w = (const float*)d_in[9];  const float* c3b = (const float*)d_in[10];
    const float* f1w = (const float*)d_in[11]; const float* f1b = (const float*)d_in[12];
    const float* f2w = (const float*)d_in[13]; const float* f2b = (const float*)d_in[14];
    const float* g1Wf = (const float*)d_in[15]; const float* g1a = (const float*)d_in[16];
    const float* g2Wf = (const float*)d_in[17]; const float* g2a = (const float*)d_in[18];
    const float* g3Wf = (const float*)d_in[19]; const float* g3a = (const float*)d_in[20];
    float* out = (float*)d_out;                      // (64,919)

    // ---- workspace bump allocator ----
    size_t off = 0;
    auto alloc = [&](size_t bytes) -> void* {
        off = (off + 255) & ~(size_t)255;
        void* p = (char*)d_ws + off;
        off += bytes;
        return p;
    };

    bf16* fc1wT = (bf16*)alloc(50880LL * 925 * 2);   // fc1_w^T
    bf16* fc2wT = (bf16*)alloc(1053LL * 919 * 2);    // fc2_w^T
    bf16* w1b   = (bf16*)alloc(320LL * 32 * 2);
    bf16* w2b   = (bf16*)alloc(480LL * 2560 * 2);
    bf16* w3b   = (bf16*)alloc(960LL * 3840 * 2);
    bf16* g1W   = (bf16*)alloc(4LL * 768 * 250 * 2);
    bf16* g2W   = (bf16*)alloc(4LL * 1000 * 100 * 2);
    bf16* g3W   = (bf16*)alloc(400LL * 128 * 2);
    bf16* nodeb = (bf16*)alloc((long long)N_NODES * 768 * 2);
    bf16* idxb  = (bf16*)alloc(64LL * N_NODES * 2);
    bf16* P1    = (bf16*)alloc(64LL * 320 * 248 * 2);
    bf16* P2    = (bf16*)alloc(64LL * 480 * 60 * 2);
    bf16* Yb    = (bf16*)alloc(64LL * 50880 * 2);
    bf16* x1    = (bf16*)alloc((long long)N_NODES * 1000 * 2);
    bf16* x2    = (bf16*)alloc((long long)N_NODES * 400 * 2);
    bf16* x3    = (bf16*)alloc((long long)N_NODES * 128 * 2);
    float* seqo = (float*)alloc(64LL * 925 * 4);
    float* strb = (float*)alloc(64LL * 128 * 4);
    bf16* catb  = (bf16*)alloc(64LL * 1053 * 2);
    float* srow = (float*)alloc((long long)N_NODES * 4);
    float* scol = (float*)alloc((long long)N_NODES * 4);
    float* den  = (float*)alloc((long long)N_NODES * 4);
    float* ebuf = (float*)alloc((long long)N_EDGES * 4);
    // big reusable scratch: im2col buffers / GAT h live here
    char* SCRB = (char*)alloc(2560LL * 15424 * 2);         // max im2col (conv2): ~79 MB
    char* SCRC = (char*)alloc(320LL * 64 * 993 * 4);       // max conv out (conv1): ~81 MB

    auto gemm = [&](const bf16* A, const bf16* B, float* C,
                    int M, int N, int K, int lda, int ldb, int ldc,
                    const float* bias, int bmode, int act) {
        dim3 grid((N + TN - 1) / TN, (M + TM - 1) / TM);
        gemm_bf16_wmma_k<<<grid, 256, 0, stream>>>(A, B, C, M, N, K, lda, ldb, ldc,
                                                   bias, bmode, act);
    };

    // ---- one-time (per call) precision conversions ----
    f32_to_bf16_T_k<<<g1d(925LL * 50880), 256, 0, stream>>>(f1w, fc1wT, 925, 50880);
    f32_to_bf16_T_k<<<g1d(919LL * 1053), 256, 0, stream>>>(f2w, fc2wT, 919, 1053);
    f32_to_bf16_k<<<g1d(320LL * 32), 256, 0, stream>>>(c1w, w1b, 320LL * 32);
    f32_to_bf16_k<<<g1d(480LL * 2560), 256, 0, stream>>>(c2w, w2b, 480LL * 2560);
    f32_to_bf16_k<<<g1d(960LL * 3840), 256, 0, stream>>>(c3w, w3b, 960LL * 3840);
    f32_to_bf16_k<<<g1d(768000), 256, 0, stream>>>(g1Wf, g1W, 768000);
    f32_to_bf16_k<<<g1d(400000), 256, 0, stream>>>(g2Wf, g2W, 400000);
    f32_to_bf16_k<<<g1d(51200), 256, 0, stream>>>(g3Wf, g3W, 51200);
    f32_to_bf16_k<<<g1d((long long)N_NODES * 768), 256, 0, stream>>>(node_in, nodeb,
                                                  (long long)N_NODES * 768);
    f32_to_bf16_k<<<g1d(64LL * N_NODES), 256, 0, stream>>>(index_in, idxb, 64LL * N_NODES);

    // ---- CNN front-end (implicit GEMM via im2col) ----
    bf16*  Bbuf = (bf16*)SCRB;
    float* Cbuf = (float*)SCRC;

    // conv1: Cin=4 K=8 Lin=1000 Lout=993 ; GEMM 320 x 63552, K=32
    im2col_k<float><<<g1d(32LL * 64 * 993), 256, 0, stream>>>(seq_in, Bbuf, 64, 4, 1000, 8, 993);
    gemm(w1b, Bbuf, Cbuf, 320, 64 * 993, 32, 32, 64 * 993, 64 * 993, c1b, 1, 0);
    relu_pool4_k<<<g1d(64LL * 320 * 248), 256, 0, stream>>>(Cbuf, P1, 64, 320, 993, 248);

    // conv2: Cin=320 K=8 Lin=248 Lout=241 ; GEMM 480 x 15424, K=2560
    im2col_k<bf16><<<g1d(2560LL * 64 * 241), 256, 0, stream>>>(P1, Bbuf, 64, 320, 248, 8, 241);
    gemm(w2b, Bbuf, Cbuf, 480, 64 * 241, 2560, 2560, 64 * 241, 64 * 241, c2b, 1, 0);
    relu_pool4_k<<<g1d(64LL * 480 * 60), 256, 0, stream>>>(Cbuf, P2, 64, 480, 241, 60);

    // conv3: Cin=480 K=8 Lin=60 Lout=53 ; GEMM 960 x 3392, K=3840
    im2col_k<bf16><<<g1d(3840LL * 64 * 53), 256, 0, stream>>>(P2, Bbuf, 64, 480, 60, 8, 53);
    gemm(w3b, Bbuf, Cbuf, 960, 64 * 53, 3840, 3840, 64 * 53, 64 * 53, c3b, 1, 0);
    pack_y_k<<<g1d(64LL * 960 * 53), 256, 0, stream>>>(Cbuf, Yb);

    // fc1: (64 x 925) = Y (64 x 50880) @ fc1_w^T ; relu
    gemm(Yb, fc1wT, seqo, 64, 925, 50880, 50880, 925, 925, f1b, 2, 1);

    // ---- GAT stack ----
    float* h   = (float*)SCRB;   // 20000 x Fmax(250) f32
    float* num = (float*)SCRC;   // 20000 x Fmax(250) f32

    auto gat_head = [&](const bf16* xin, int Fin, const bf16* W, const float* a,
                        int F, bf16* xout, int ldo, int offo) {
        gemm(xin, W, h, N_NODES, F, Fin, Fin, F, F, nullptr, 0, 0);
        zero_f32_k<<<g1d(N_NODES), 256, 0, stream>>>(den, N_NODES);
        zero_f32_k<<<g1d((long long)N_NODES * F), 256, 0, stream>>>(num, (long long)N_NODES * F);
        gat_scores_k<<<g1d((long long)N_NODES * 32), 256, 0, stream>>>(h, a, F, srow, scol, N_NODES);
        gat_edge_k<<<g1d(N_EDGES), 256, 0, stream>>>(e_row, e_col, srow, scol, ebuf, den, N_EDGES);
        gat_scatter_k<<<g1d((long long)N_EDGES * F), 256, 0, stream>>>(e_row, e_col, ebuf, h, num,
                                                                       N_EDGES, F);
        gat_finalize_k<<<g1d((long long)N_NODES * F), 256, 0, stream>>>(num, den, xout,
                                                                        N_NODES, F, ldo, offo);
    };

    // layer 1: 768 -> 4 x 250 (concat 1000)
    for (int hd = 0; hd < 4; ++hd)
        gat_head(nodeb, 768, g1W + (long long)hd * 768 * 250, g1a + hd * 500,
                 250, x1, 1000, hd * 250);
    // layer 2: 1000 -> 4 x 100 (concat 400)
    for (int hd = 0; hd < 4; ++hd)
        gat_head(x1, 1000, g2W + (long long)hd * 1000 * 100, g2a + hd * 200,
                 100, x2, 400, hd * 100);
    // layer 3: 400 -> 128
    gat_head(x2, 400, g3W, g3a, 128, x3, 128, 0);

    // struct = index_input @ x3 : (64 x 128), K = 20000
    gemm(idxb, x3, strb, 64, 128, N_NODES, N_NODES, 128, 128, nullptr, 0, 0);

    // concat + relu -> bf16, then fc2 + sigmoid -> d_out
    pack_cat_k<<<g1d(64LL * 1053), 256, 0, stream>>>(seqo, strb, catb);
    gemm(catb, fc2wT, out, 64, 919, 1053, 1053, 919, 919, f2b, 2, 2);
}
